// NTXentLoss_8478265442622
// MI455X (gfx1250) — matrix operand order repaired and checked by reading, not compiled, not run on hardware
//
#include <hip/hip_runtime.h>

// ---------- types ----------
typedef __attribute__((ext_vector_type(16))) __bf16 v16bf;
typedef __attribute__((ext_vector_type(8)))  float  v8f;

union Frag16 { v16bf v; uint4 u[2]; };   // 32 bytes: 16 bf16 == two 16B chunks

#define NROWS 4096   // 2*B
#define DDIM  512
#define INVT  2.0f   // 1/temperature
#define FMAX  2.0f   // logits are cosine/T in [-2, 2] -> fixed logsumexp shift

__device__ __forceinline__ unsigned short f32_to_bf16(float f) {
    unsigned int u = __builtin_bit_cast(unsigned int, f);
    u += 0x7FFFu + ((u >> 16) & 1u);      // round to nearest even
    return (unsigned short)(u >> 16);
}

// ---------- kernel 1: row-normalize z = concat(proj_i, proj_j) into bf16 ----------
// grid 512 x 256 threads: 8 waves/block, one wave (32 lanes) per row.
__global__ __launch_bounds__(256) void ntx_normalize(const float* __restrict__ pi,
                                                     const float* __restrict__ pj,
                                                     unsigned short* __restrict__ zn) {
    const int lane = threadIdx.x & 31;
    const int wave = threadIdx.x >> 5;
    const int row  = blockIdx.x * 8 + wave;
    const float* src = (row < NROWS / 2) ? (pi + (size_t)row * DDIM)
                                         : (pj + (size_t)(row - NROWS / 2) * DDIM);
    const float4* s4 = (const float4*)src;

    float4 x[4];
    float ss = 0.f;
#pragma unroll
    for (int i = 0; i < 4; ++i) {
        x[i] = s4[lane + 32 * i];
        ss += x[i].x * x[i].x + x[i].y * x[i].y + x[i].z * x[i].z + x[i].w * x[i].w;
    }
#pragma unroll
    for (int m = 16; m >= 1; m >>= 1) ss += __shfl_xor(ss, m, 32);

    const float inv = rsqrtf(fmaxf(ss, 1e-24f));   // eps never binds for N(0,1) rows

    ushort4* dst = (ushort4*)(zn + (size_t)row * DDIM);
#pragma unroll
    for (int i = 0; i < 4; ++i) {
        ushort4 o;
        o.x = f32_to_bf16(x[i].x * inv);
        o.y = f32_to_bf16(x[i].y * inv);
        o.z = f32_to_bf16(x[i].z * inv);
        o.w = f32_to_bf16(x[i].w * inv);
        dst[lane + 32 * i] = o;
    }
}

// ---------- helpers for kernel 2 ----------
// B-fragment tile: 16 k-steps, per lane two contiguous 16B chunks (K [half*16, half*16+16))
__device__ __forceinline__ void load_btile(const uint4* __restrict__ brp, Frag16 b[16]) {
#pragma unroll
    for (int kt = 0; kt < 16; ++kt) {
        b[kt].u[0] = brp[kt * 4];
        b[kt].u[1] = brp[kt * 4 + 1];
    }
}

__device__ __forceinline__ void compute_tile(const Frag16 a[16], const Frag16 b[16], v8f& acc) {
#pragma unroll
    for (int kt = 0; kt < 16; ++kt) {
        acc = __builtin_amdgcn_wmma_f32_16x16x32_bf16(
            false, a[kt].v, false, b[kt].v, (short)0, acc, false, false);
    }
}

// ---------- kernel 2: fused Gram-GEMM + partial logsumexp ----------
// grid = 256 blocks: blockIdx = (row_block[0..31] << 3) | chunk[0..7]
// 8 waves/block, each wave owns 16 rows and sweeps a 512-column chunk.
// B tiles are double-buffered in registers: tile t+1's 32 loads are issued
// before tile t's 16-WMMA chain, so L2 latency overlaps matrix compute.
__global__ __launch_bounds__(256) void ntx_simlse(const unsigned short* __restrict__ zn,
                                                  float* __restrict__ partial_s,
                                                  float* __restrict__ pos) {
    const int lane = threadIdx.x & 31;
    const int wave = threadIdx.x >> 5;
    const int rb   = blockIdx.x >> 3;
    const int ch   = blockIdx.x & 7;
    const int rt   = rb * 128 + wave * 16;

    const uint4* z4 = (const uint4*)zn;        // 8 bf16 per uint4, row stride = 64 uint4
    const int half = lane >> 4;                // 0: lanes 0-15, 1: lanes 16-31
    const int l15  = lane & 15;

    // A fragments: lane holds row (rt+l15),
    // K contiguous [half*16, half*16+16) (same order as B). Loaded once,
    // register-resident for the whole column sweep (128 VGPRs).
    const uint4* arp = z4 + (size_t)(rt + l15) * 64 + half * 2;
    Frag16 a[16];
#pragma unroll
    for (int kt = 0; kt < 16; ++kt) {
        a[kt].u[0] = arp[kt * 4];
        a[kt].u[1] = arp[kt * 4 + 1];
    }

    float ssum[8];
#pragma unroll
    for (int v = 0; v < 8; ++v) ssum[v] = 0.f;
    const int rglob_base = rt + 8 * half;      // C layout: M = v + 8*half

    const int cbase0 = ch * 512;
    // B = zn^T: column n of B is row (ctbase+n) of zn; lane holds col l15,
    // K contiguous [half*16, half*16+16). Tile->tile stride: 16 rows = 1024 uint4.
    const uint4* brp = z4 + (size_t)(cbase0 + l15) * 64 + half * 2;

    auto epilogue = [&](const v8f& acc, int ctbase) {
        const int cglob = ctbase + l15;
#pragma unroll
        for (int v = 0; v < 8; ++v) {
            const int rglob = rglob_base + v;
            const float x = acc[v] * INVT;               // logit in [-2, 2]
            float e = __expf(x - FMAX);                  // fixed-shift stable exp
            if (cglob == rglob) e = 0.f;                 // diagonal masked (-inf)
            ssum[v] += e;
            if (cglob == (rglob ^ 2048)) pos[rglob] = x; // positive pair logit
        }
    };

    Frag16 b0[16], b1[16];
    load_btile(brp, b0);                       // prime the pipeline
    for (int ct8 = 0; ct8 < 32; ct8 += 2) {
        load_btile(brp + 1024, b1);            // prefetch tile ct8+1
        v8f acc = {};
        compute_tile(a, b0, acc);
        epilogue(acc, cbase0 + ct8 * 16);

        // prefetch tile ct8+2 (last iteration over-reads one tile past the
        // chunk -- still inside the workspace, values never consumed)
        load_btile(brp + 2048, b0);
        v8f acc2 = {};
        compute_tile(a, b1, acc2);
        epilogue(acc2, cbase0 + ct8 * 16 + 16);

        brp += 2048;
    }

    // reduce sums across the 16 lanes of each half (xor masks stay in-half)
#pragma unroll
    for (int m = 1; m < 16; m <<= 1) {
#pragma unroll
        for (int v = 0; v < 8; ++v) ssum[v] += __shfl_xor(ssum[v], m, 32);
    }
    if (l15 == 0) {
#pragma unroll
        for (int v = 0; v < 8; ++v)
            partial_s[(size_t)(rglob_base + v) * 8 + ch] = ssum[v];
    }
}

// ---------- kernel 3: combine chunks, per-row loss, mean ----------
__global__ __launch_bounds__(1024) void ntx_finalize(const float* __restrict__ partial_s,
                                                     const float* __restrict__ pos,
                                                     float* __restrict__ out) {
    __shared__ float red[1024];
    const int t = threadIdx.x;
    float acc = 0.f;
    for (int r = t; r < NROWS; r += 1024) {
        float S = 0.f;
#pragma unroll
        for (int c = 0; c < 8; ++c) S += partial_s[(size_t)r * 8 + c];
        acc += (FMAX + logf(S)) - pos[r];    // lse - positive logit
    }
    red[t] = acc;
    __syncthreads();
    for (int s = 512; s > 0; s >>= 1) {
        if (t < s) red[t] += red[t + s];
        __syncthreads();
    }
    if (t == 0) out[0] = red[0] * (1.0f / (float)NROWS);
}

extern "C" void kernel_launch(void* const* d_in, const int* in_sizes, int n_in,
                              void* d_out, int out_size, void* d_ws, size_t ws_size,
                              hipStream_t stream) {
    (void)in_sizes; (void)n_in; (void)out_size; (void)ws_size;
    const float* pi = (const float*)d_in[0];
    const float* pj = (const float*)d_in[1];

    // workspace layout
    unsigned short* zn = (unsigned short*)d_ws;                            // 4096*512*2 = 4 MB
    float* partial_s   = (float*)((char*)d_ws + (size_t)NROWS * DDIM * 2); // 4096*8*4 = 128 KB
    float* pos         = partial_s + (size_t)NROWS * 8;                    // 4096*4  = 16 KB

    ntx_normalize<<<NROWS / 8, 256, 0, stream>>>(pi, pj, zn);
    ntx_simlse<<<256, 256, 0, stream>>>(zn, partial_s, pos);
    ntx_finalize<<<1, 1024, 0, stream>>>(partial_s, pos, (float*)d_out);
}
